// Capsule_65919158059332
// MI455X (gfx1250) — matrix operand order repaired and checked by reading, not compiled, not run on hardware
//
#include <hip/hip_runtime.h>
#include <hip/hip_bf16.h>

// ---------------------------------------------------------------------------
// Types
// ---------------------------------------------------------------------------
typedef __bf16           v16bf __attribute__((ext_vector_type(16)));
typedef float            v8f   __attribute__((ext_vector_type(8)));
typedef unsigned short   u16x8 __attribute__((ext_vector_type(8)));
typedef unsigned short   u16x16 __attribute__((ext_vector_type(16)));

#define B_SZ  64
#define IN_SZ 1024
#define ID_SZ 256
#define NC    32
#define DC    64
#define M_SZ  (B_SZ * IN_SZ)   // 65536
#define N_SZ  (NC * DC)        // 2048
#define IBLK  128              // i's per routing block
#define CHK   64               // i's per routing chunk (L2-friendly re-read)

__device__ __forceinline__ unsigned short f2bf(float f) {
    union { float f; unsigned u; } v; v.f = f;
    unsigned u = v.u;
    u += 0x7FFFu + ((u >> 16) & 1u);   // round-to-nearest-even
    return (unsigned short)(u >> 16);
}
__device__ __forceinline__ float bf2f(unsigned short h) {
    union { unsigned u; float f; } v; v.u = ((unsigned)h) << 16;
    return v.f;
}

// ---------------------------------------------------------------------------
// K0: f32 -> bf16 conversion (vectorized, 4 elems/thread)
// ---------------------------------------------------------------------------
__global__ __launch_bounds__(256) void k_f32_to_bf16(const float* __restrict__ in,
                                                     unsigned short* __restrict__ out,
                                                     int n) {
    int idx = (blockIdx.x * 256 + threadIdx.x) * 4;
    if (idx + 3 < n) {
        float4 v = *(const float4*)(in + idx);
        out[idx + 0] = f2bf(v.x);
        out[idx + 1] = f2bf(v.y);
        out[idx + 2] = f2bf(v.z);
        out[idx + 3] = f2bf(v.w);
    }
}

__global__ __launch_bounds__(256) void k_zero_f32(float* __restrict__ p, int n) {
    int idx = blockIdx.x * 256 + threadIdx.x;
    if (idx < n) p[idx] = 0.0f;
}

// ---------------------------------------------------------------------------
// K1: bf16 WMMA GEMM + fused iteration-0 column sums.
//   u_hat[b,n,i,d] = sum_k A[b*IN+i,k] * W[n*DC+d,k]   (NT GEMM, K-major)
//   Workgroup tile 128x128, 8 waves (2M x 4N), wave tile 64x32 = 4x2 WMMA.
//   u_hat stored bf16 directly in [B,NC,IN,DC]; column sums Sum_i u_hat
//   reduced in LDS (ds_add_f32) then one global atomicAdd per column.
//   (F.normalize is scale-invariant, so raw sums == mean for iteration 0.)
// ---------------------------------------------------------------------------
__global__ __launch_bounds__(256) void k_gemm_wmma(const unsigned short* __restrict__ A,
                                                   const unsigned short* __restrict__ Wt,
                                                   unsigned short* __restrict__ U,
                                                   float* __restrict__ o0raw) {
    __shared__ unsigned short As[128 * 32];   // 8 KB
    __shared__ unsigned short Bs[128 * 32];   // 8 KB
    __shared__ float colsum[128];             // per-block Sum_i of each output col

    const int tid  = threadIdx.x;
    const int lane = tid & 31;
    const int wave = tid >> 5;
    const int wm   = wave >> 2;      // 0..1  (M direction)
    const int wn   = wave & 3;       // 0..3  (N direction)
    const int m0   = blockIdx.x * 128;
    const int n0   = blockIdx.y * 128;

    if (tid < 128) colsum[tid] = 0.0f;   // ordered vs. use by in-loop barriers

    v8f acc[4][2];
#pragma unroll
    for (int mi = 0; mi < 4; ++mi)
#pragma unroll
        for (int ni = 0; ni < 2; ++ni)
            acc[mi][ni] = (v8f){0.f,0.f,0.f,0.f,0.f,0.f,0.f,0.f};

    // cooperative tile load mapping: 256 threads, 128 rows x 32 cols bf16
    const int row = tid >> 1;            // 0..127
    const int col = (tid & 1) * 16;      // 0 or 16

    // wave32 WMMA fragment lane mapping (CDNA5 ISA 7.12.2):
    //  A (16x32 bf16): lane L -> row M=L&15; e<8 -> K=kA+e, e>=8 -> K=kA+16+(e-8); kA=(L>>4)*8
    //  B (32x16 bf16): lane L -> col N=L&15; e<8 -> K=kB+e, e>=8 -> K=kB+16+(e-8); kB=(L>>4)*8
    const int fr = lane & 15;
    const int kA = (lane >> 4) * 8;
    const int kB = (lane >> 4) * 8;

    for (int k0 = 0; k0 < ID_SZ; k0 += 32) {
        const u16x8* ga = (const u16x8*)(A  + (size_t)(m0 + row) * ID_SZ + k0 + col);
        const u16x8* gb = (const u16x8*)(Wt + (size_t)(n0 + row) * ID_SZ + k0 + col);
        *(u16x8*)(&As[row * 32 + col])     = ga[0];
        *(u16x8*)(&As[row * 32 + col + 8]) = ga[1];
        *(u16x8*)(&Bs[row * 32 + col])     = gb[0];
        *(u16x8*)(&Bs[row * 32 + col + 8]) = gb[1];
        if (k0 + 32 < ID_SZ) {  // next K tile -> global_prefetch_b8
            __builtin_prefetch(A  + (size_t)(m0 + row) * ID_SZ + k0 + 32 + col, 0, 1);
            __builtin_prefetch(Wt + (size_t)(n0 + row) * ID_SZ + k0 + 32 + col, 0, 1);
        }
        __syncthreads();

        v16bf bfrag[2];
#pragma unroll
        for (int ni = 0; ni < 2; ++ni) {
            const unsigned short* p = &Bs[(wn * 32 + ni * 16 + fr) * 32 + kB];
            u16x8 lo = *(const u16x8*)p;
            u16x8 hi = *(const u16x8*)(p + 16);
            u16x16 full = __builtin_shufflevector(lo, hi, 0,1,2,3,4,5,6,7,8,9,10,11,12,13,14,15);
            bfrag[ni] = __builtin_bit_cast(v16bf, full);
        }
#pragma unroll
        for (int mi = 0; mi < 4; ++mi) {
            const unsigned short* p = &As[(wm * 64 + mi * 16 + fr) * 32 + kA];
            u16x8 lo = *(const u16x8*)p;         // K = kA .. kA+7
            u16x8 hi = *(const u16x8*)(p + 16);  // K = kA+16 .. kA+23
            u16x16 full = __builtin_shufflevector(lo, hi, 0,1,2,3,4,5,6,7,8,9,10,11,12,13,14,15);
            v16bf afrag = __builtin_bit_cast(v16bf, full);
#pragma unroll
            for (int ni = 0; ni < 2; ++ni) {
                acc[mi][ni] = __builtin_amdgcn_wmma_f32_16x16x32_bf16(
                    false, afrag, false, bfrag[ni], (short)0, acc[mi][ni], false, false);
            }
        }
        __syncthreads();
    }

    // Epilogue 1: scatter-store bf16 u_hat into [B,NC,IN,DC].
    // C tile lane map: lane L -> col=L&15, elem v -> row = v + (L>>4)*8
#pragma unroll
    for (int mi = 0; mi < 4; ++mi) {
#pragma unroll
        for (int ni = 0; ni < 2; ++ni) {
            const int nn   = n0 + wn * 32 + ni * 16 + (lane & 15);
            const int ncap = nn >> 6;
            const int d    = nn & 63;
            const int mb   = m0 + wm * 64 + mi * 16 + ((lane >> 4) * 8);
#pragma unroll
            for (int v = 0; v < 8; ++v) {
                const int m = mb + v;
                const int b = m >> 10;
                const int i = m & 1023;
                U[((((size_t)b * NC + ncap) * IN_SZ + i) * DC) + d] = f2bf(acc[mi][ni][v]);
            }
        }
    }

    // Epilogue 2: fused iteration-0 sums.  Per lane: sum its 4 (mi) x 8 (v)
    // rows for each of its 2 columns, LDS-reduce (ds_add_f32), then one
    // global atomicAdd per column per block.
#pragma unroll
    for (int ni = 0; ni < 2; ++ni) {
        float s = 0.0f;
#pragma unroll
        for (int mi = 0; mi < 4; ++mi)
#pragma unroll
            for (int v = 0; v < 8; ++v) s += acc[mi][ni][v];
        atomicAdd(&colsum[wn * 32 + ni * 16 + (lane & 15)], s);
    }
    __syncthreads();
    if (tid < 128) {
        const int nn = n0 + tid;
        const int b  = m0 >> 10;                 // one batch per 128-row block
        atomicAdd(o0raw + ((size_t)b * NC + (nn >> 6)) * DC + (nn & 63), colsum[tid]);
    }
}

// ---------------------------------------------------------------------------
// K3: fused routing pass, chunked for L2 reuse. One block per (b, IBLK i's).
//   Per 64-i chunk: phase A computes logits into LDS (padded stride 33),
//   per-i softmax in place, phase B immediately re-reads the same 256 KB
//   u_hat slab (L2-hot) accumulating into chunk-invariant register acc[8].
//   One set of global f32 atomics per block at the end.
// ---------------------------------------------------------------------------
__global__ __launch_bounds__(256) void k_route(const unsigned short* __restrict__ U,
                                               const float* __restrict__ ohat,
                                               float* __restrict__ onext) {
    const int b  = blockIdx.x;
    const int i0 = blockIdx.y * IBLK;
    const int t  = threadIdx.x;

    __shared__ float oh[NC * DC];      // 8 KB   (o-hat for this batch)
    __shared__ float lg[CHK * 33];     // 8.25 KB logits -> c, padded stride

#pragma unroll
    for (int j = 0; j < 8; ++j) oh[t + j * 256] = ohat[(size_t)b * NC * DC + t + j * 256];

    const int il = t >> 2;             // phase A: i within chunk (0..63)
    const int q  = t & 3;              // phase A: n octet (0..3)
    const int pn = t >> 3;             // phase B: capsule n (0..31)
    const int pd = (t & 7) * 8;        // phase B: d base
    float acc[8] = {0.f,0.f,0.f,0.f,0.f,0.f,0.f,0.f};

    __syncthreads();

    for (int ch = 0; ch < IBLK / CHK; ++ch) {
        const int ic = i0 + ch * CHK;

        // phase A: logits[i][n] = <u_hat[b,n,i,:], oh[n,:]>
#pragma unroll
        for (int k = 0; k < 8; ++k) {
            const int n = q * 8 + k;
            const unsigned short* up = U + (((size_t)b * NC + n) * IN_SZ + ic + il) * DC;
            float dot = 0.0f;
#pragma unroll
            for (int d = 0; d < DC; d += 8) {
                u16x8 v = *(const u16x8*)(up + d);
#pragma unroll
                for (int j = 0; j < 8; ++j) dot += bf2f(v[j]) * oh[n * DC + d + j];
            }
            lg[il * 33 + n] = dot;
        }
        __syncthreads();

        // softmax over n, per i (64 threads, in place)
        if (t < CHK) {
            float* L = &lg[t * 33];
            float mx = L[0];
#pragma unroll
            for (int n = 1; n < NC; ++n) mx = fmaxf(mx, L[n]);
            float s = 0.0f;
#pragma unroll
            for (int n = 0; n < NC; ++n) { float e = __expf(L[n] - mx); L[n] = e; s += e; }
            const float inv = 1.0f / s;
#pragma unroll
            for (int n = 0; n < NC; ++n) L[n] *= inv;
        }
        __syncthreads();

        // phase B: acc[d] += sum_i c[n,i] * u_hat[b,n,i,d]  (slab is L2-hot)
        const unsigned short* ub = U + (((size_t)b * NC + pn) * IN_SZ + ic) * DC + pd;
        for (int ii = 0; ii < CHK; ++ii) {
            u16x8 v = *(const u16x8*)(ub + (size_t)ii * DC);
            const float c = lg[ii * 33 + pn];
#pragma unroll
            for (int j = 0; j < 8; ++j) acc[j] += c * bf2f(v[j]);
        }
        __syncthreads();   // protect lg before next chunk overwrites it
    }

    float* op = onext + ((size_t)b * NC + pn) * DC + pd;
#pragma unroll
    for (int j = 0; j < 8; ++j) atomicAdd(op + j, acc[j]);
}

// ---------------------------------------------------------------------------
// K4: per-(b,n) row: normalize (mode 0) or squash (mode 1).
// ---------------------------------------------------------------------------
__global__ __launch_bounds__(64) void k_finish(const float* __restrict__ o,
                                               float* __restrict__ out,
                                               int squash) {
    const int bn = blockIdx.x, t = threadIdx.x;
    const float v = o[(size_t)bn * 64 + t];
    __shared__ float red[64];
    red[t] = v * v;
    __syncthreads();
    for (int s = 32; s > 0; s >>= 1) {
        if (t < s) red[t] += red[t + s];
        __syncthreads();
    }
    const float ss = red[0];
    float r;
    if (squash) {
        const float sq = ss + 1e-7f;
        r = (sqrtf(sq) / (0.5f + sq)) * v;
    } else {
        r = v / fmaxf(sqrtf(ss), 1e-12f);
    }
    out[(size_t)bn * 64 + t] = r;
}

// ---------------------------------------------------------------------------
// Host orchestration
// ---------------------------------------------------------------------------
extern "C" void kernel_launch(void* const* d_in, const int* in_sizes, int n_in,
                              void* d_out, int out_size, void* d_ws, size_t ws_size,
                              hipStream_t stream) {
    const float* u_vecs = (const float*)d_in[0];   // [64,1024,256]
    const float* W      = (const float*)d_in[1];   // [2048,256]
    float* out          = (float*)d_out;           // [64,32,64]

    char* ws = (char*)d_ws;
    size_t off = 0;
    unsigned short* A_bf = (unsigned short*)(ws + off); off += (size_t)M_SZ * ID_SZ * 2;  // 32 MB
    unsigned short* W_bf = (unsigned short*)(ws + off); off += (size_t)N_SZ * ID_SZ * 2;  // 1 MB
    unsigned short* Uh   = (unsigned short*)(ws + off); off += (size_t)M_SZ * N_SZ * 2;   // 256 MB
    float* oA = (float*)(ws + off); off += (size_t)B_SZ * NC * DC * 4;                    // accum
    float* oB = (float*)(ws + off); off += (size_t)B_SZ * NC * DC * 4;                    // o-hat

    const int nO = B_SZ * NC * DC;  // 131072

    // convert inputs to bf16
    k_f32_to_bf16<<<(M_SZ * ID_SZ) / 1024, 256, 0, stream>>>(u_vecs, A_bf, M_SZ * ID_SZ);
    k_f32_to_bf16<<<(N_SZ * ID_SZ) / 1024, 256, 0, stream>>>(W, W_bf, N_SZ * ID_SZ);

    // u_hat GEMM (WMMA bf16) + fused iteration-0 column sums into oA
    k_zero_f32<<<(nO + 255) / 256, 256, 0, stream>>>(oA, nO);
    k_gemm_wmma<<<dim3(M_SZ / 128, N_SZ / 128), 256, 0, stream>>>(A_bf, W_bf, Uh, oA);

    // iteration 0: normalize raw sums (normalize is scale-invariant) -> oB
    k_finish<<<B_SZ * NC, 64, 0, stream>>>(oA, oB, 0);

    // iteration 1: fused logits/softmax/weighted-sum, then normalize
    k_zero_f32<<<(nO + 255) / 256, 256, 0, stream>>>(oA, nO);
    k_route<<<dim3(B_SZ, IN_SZ / IBLK), 256, 0, stream>>>(Uh, oB, oA);
    k_finish<<<B_SZ * NC, 64, 0, stream>>>(oA, oB, 0);

    // iteration 2 + final squash -> output
    k_zero_f32<<<(nO + 255) / 256, 256, 0, stream>>>(oA, nO);
    k_route<<<dim3(B_SZ, IN_SZ / IBLK), 256, 0, stream>>>(Uh, oB, oA);
    k_finish<<<B_SZ * NC, 64, 0, stream>>>(oA, out, 1);
}
